// Qwen3VLVisionBlock_27857157882050
// MI455X (gfx1250) — compile-verified, matrix operand order checked by reading, not run on hardware
//
#include <hip/hip_runtime.h>

#define L_SEQ 4096
#define HID 1152
#define NH 16
#define HD 72
#define HDQ 96    // HD padded to K-multiple of 32 for QK^T
#define HDV 80    // HD padded to N-multiple of 16 for PV
#define FF 4304
#define FFP 4352  // FF padded to multiple of 128
#define QKV_N 3456

typedef __attribute__((ext_vector_type(16))) __bf16 v16bf;
typedef __attribute__((ext_vector_type(8)))  float  v8f;

union Frag16 { v16bf v; uint4 u[2]; };

__device__ __forceinline__ unsigned int f2bf(float f) {
  unsigned int u = __float_as_uint(f);
  return (u + 0x7FFFu + ((u >> 16) & 1u)) >> 16;   // round-to-nearest-even
}
__device__ __forceinline__ unsigned int pk2bf(float lo, float hi) {
  return f2bf(lo) | (f2bf(hi) << 16);
}
__device__ __forceinline__ float gelu_tanh(float x) {
  float x3 = x * x * x;
  return 0.5f * x * (1.f + tanhf(0.7978845608028654f * (x + 0.044715f * x3)));
}

// ---- gfx1250 async global->LDS copy (16B per lane), tracked by ASYNCcnt ----
// LDS operand: low 32 bits of the flat shared-memory pointer == LDS byte offset
// (flat->LDS mapping truncates to addr[31:0], ISA 10.2).
__device__ __forceinline__ void async_copy16(const void* lds, const void* gptr) {
  unsigned l = (unsigned)(size_t)lds;
  asm volatile("global_load_async_to_lds_b128 %0, %1, off"
               :: "v"(l), "v"(gptr) : "memory");
}
__device__ __forceinline__ void wait_async0() {
  asm volatile("s_wait_asynccnt 0" ::: "memory");
}

// ---------------------------------------------------------------- LayerNorm
__global__ __launch_bounds__(256)
void ln_kernel(const float* __restrict__ x, const float* __restrict__ sc,
               const float* __restrict__ bi, float* __restrict__ y) {
  __shared__ float red[256];
  const int row = blockIdx.x, tid = threadIdx.x;
  const float* xr = x + (size_t)row * HID;
  float s = 0.f;
  for (int i = tid; i < HID; i += 256) s += xr[i];
  red[tid] = s; __syncthreads();
  for (int o = 128; o > 0; o >>= 1) { if (tid < o) red[tid] += red[tid + o]; __syncthreads(); }
  const float mu = red[0] * (1.f / HID);
  __syncthreads();
  float s2 = 0.f;
  for (int i = tid; i < HID; i += 256) { float d = xr[i] - mu; s2 += d * d; }
  red[tid] = s2; __syncthreads();
  for (int o = 128; o > 0; o >>= 1) { if (tid < o) red[tid] += red[tid + o]; __syncthreads(); }
  const float inv = rsqrtf(red[0] * (1.f / HID) + 1e-6f);
  for (int i = tid; i < HID; i += 256)
    y[(size_t)row * HID + i] = (xr[i] - mu) * inv * sc[i] + bi[i];
}

// ------------------------------------------- Weight transpose+convert to bf16
__global__ __launch_bounds__(256)
void convT_kernel(const float* __restrict__ W, unsigned short* __restrict__ Wt,
                  int K, int N, int Kp, int Np) {
  long long idx = (long long)blockIdx.x * 256 + threadIdx.x;
  if (idx >= (long long)Np * Kp) return;
  int n = (int)(idx / Kp), k = (int)(idx % Kp);
  unsigned short v = 0;
  if (n < N && k < K) v = (unsigned short)f2bf(W[(size_t)k * N + n]);
  Wt[(size_t)n * Kp + k] = v;
}

// ------------------------------------------------------- bf16 WMMA GEMM
// C[M][ldc] = act(A[M][lda(f32)] x Wt^T + bias) (+res). M = 128*gridDim.y.
template <int ACT, int RES>
__global__ __launch_bounds__(256)
void gemm_bf16(const float* __restrict__ A, int lda,
               const unsigned short* __restrict__ Bt,
               const float* __restrict__ bias,
               const float* __restrict__ res, int ldr,
               float* __restrict__ C, int ldc, int Nvalid, int Kp) {
  __shared__ alignas(16) unsigned short lsA[128 * 72];
  __shared__ alignas(16) unsigned short lsB[128 * 72];
  const int tid = threadIdx.x, lane = tid & 31, w = tid >> 5;
  const int wm = w & 3, wn = w >> 2;
  const int m0 = blockIdx.y * 128, n0 = blockIdx.x * 128;
  const int cn = lane & 15, kb8 = (lane >> 4) << 3;

  v8f acc[2][4];
  for (int i = 0; i < 2; i++)
    for (int j = 0; j < 4; j++)
      for (int r = 0; r < 8; r++) acc[i][j][r] = 0.f;

  for (int kc = 0; kc < Kp; kc += 64) {
    // Bt tile: pure bf16 copy -> async DMA into LDS (128x64)
    #pragma unroll
    for (int i = 0; i < 4; i++) {
      int f = tid + i * 256;
      int r = f >> 3, c8 = (f & 7) << 3;
      async_copy16(&lsB[r * 72 + c8], Bt + (size_t)(n0 + r) * Kp + kc + c8);
    }
    // A tile: f32 -> bf16 convert while staging (128x64)
    #pragma unroll
    for (int i = 0; i < 8; i++) {
      int f = tid + i * 256;
      int r = f >> 4, c4 = (f & 15) << 2;
      const float4 av = *(const float4*)(A + (size_t)(m0 + r) * lda + kc + c4);
      unsigned int* dst = (unsigned int*)&lsA[r * 72 + c4];
      dst[0] = pk2bf(av.x, av.y);
      dst[1] = pk2bf(av.z, av.w);
    }
    if (kc + 64 < Kp)  // prefetch next A chunk into L2/WGP$
      __builtin_prefetch(A + (size_t)(m0 + (tid >> 1)) * lda + kc + 64, 0, 0);
    wait_async0();
    __syncthreads();
    #pragma unroll
    for (int ks = 0; ks < 64; ks += 32) {
      Frag16 a0, a1, b;
      int arow0 = (wm * 32 + cn) * 72 + ks + kb8;
      a0.u[0] = *(const uint4*)&lsA[arow0];
      a0.u[1] = *(const uint4*)&lsA[arow0 + 16];
      int arow1 = arow0 + 16 * 72;
      a1.u[0] = *(const uint4*)&lsA[arow1];
      a1.u[1] = *(const uint4*)&lsA[arow1 + 16];
      #pragma unroll
      for (int j = 0; j < 4; j++) {
        int brow = (wn * 64 + j * 16 + cn) * 72 + ks + kb8;
        b.u[0] = *(const uint4*)&lsB[brow];
        b.u[1] = *(const uint4*)&lsB[brow + 16];
        acc[0][j] = __builtin_amdgcn_wmma_f32_16x16x32_bf16(
            false, a0.v, false, b.v, (short)0, acc[0][j], false, false);
        acc[1][j] = __builtin_amdgcn_wmma_f32_16x16x32_bf16(
            false, a1.v, false, b.v, (short)0, acc[1][j], false, false);
      }
    }
    __syncthreads();
  }

  const int hi = lane >> 4;
  #pragma unroll
  for (int i = 0; i < 2; i++)
    #pragma unroll
    for (int j = 0; j < 4; j++) {
      int col = n0 + wn * 64 + j * 16 + cn;
      float bv = (col < Nvalid) ? bias[col] : 0.f;
      #pragma unroll
      for (int r = 0; r < 8; r++) {
        int row = m0 + wm * 32 + i * 16 + r + 8 * hi;
        float vv = acc[i][j][r] + bv;
        if (ACT == 1) vv = gelu_tanh(vv);
        if (col >= Nvalid) vv = 0.f;
        if (RES) { if (col < Nvalid) vv += res[(size_t)row * ldr + col]; }
        if (col < ldc) C[(size_t)row * ldc + col] = vv;
      }
    }
}

// ---------------------------------------------------------------- RoPE + pack
// q/k -> [L][NH][HDQ] bf16 (zero-padded); v -> TRANSPOSED [NH][HDV][L] bf16
__global__ __launch_bounds__(256)
void rope_kernel(const float* __restrict__ qkv, const float* __restrict__ rot,
                 unsigned short* __restrict__ qp, unsigned short* __restrict__ kp,
                 unsigned short* __restrict__ vp) {
  int idx = blockIdx.x * 256 + threadIdx.x;       // L*NH*HDQ threads
  int d = idx % HDQ;
  int h = (idx / HDQ) & (NH - 1);
  int l = idx / (HDQ * NH);
  size_t po = (size_t)l * (NH * HDQ) + h * HDQ + d;
  if (d >= HD) {                                   // zero pads
    qp[po] = 0; kp[po] = 0;
    if (d < HDV) vp[(size_t)h * (HDV * L_SEQ) + (size_t)d * L_SEQ + l] = 0;
    return;
  }
  float c = cosf(rot[(size_t)l * HD + d]);
  float s = sinf(rot[(size_t)l * HD + d]);
  size_t base = (size_t)l * (3 * HID) + h * HD;
  float qv = qkv[base + d];
  float kv = qkv[base + HID + d];
  float vv = qkv[base + 2 * HID + d];
  int dd = (d < HD / 2) ? d + HD / 2 : d - HD / 2;
  float qo = qkv[base + dd];
  float ko = qkv[base + HID + dd];
  float rq = (d < HD / 2) ? -qo : qo;
  float rk = (d < HD / 2) ? -ko : ko;
  qp[po] = (unsigned short)f2bf(qv * c + rq * s);
  kp[po] = (unsigned short)f2bf(kv * c + rk * s);
  vp[(size_t)h * (HDV * L_SEQ) + (size_t)d * L_SEQ + l] = (unsigned short)f2bf(vv);
}

// ------------------------------------------- flash attention (block-diagonal)
// grid (L/128, NH); 8 waves, each owns 16 query rows, full 32-key step.
__global__ __launch_bounds__(256)
void attn_kernel(const unsigned short* __restrict__ qp,
                 const unsigned short* __restrict__ kp,
                 const unsigned short* __restrict__ vp,
                 const int* __restrict__ cu, int nseg,
                 float* __restrict__ out) {
  __shared__ alignas(16) unsigned short kt[32 * HDQ];   // [key][hd]
  __shared__ alignas(16) unsigned short vt[HDV * 32];   // [hd][key]
  __shared__ alignas(16) unsigned short pb[8 * 16 * 32];// per-wave P staging
  const int tid = threadIdx.x, lane = tid & 31, w = tid >> 5;
  const int h = blockIdx.y;
  const int qblk = blockIdx.x * 128;
  const int qrow0 = qblk + w * 16;
  const int cn = lane & 15, kb8 = (lane >> 4) << 3, hi = lane >> 4;

  int kbeg = 0, kend = L_SEQ;
  for (int s = 0; s < nseg; s++) {
    int a = cu[s], b = cu[s + 1];
    if (a <= qblk && qblk < b) { kbeg = a; kend = b; }
  }

  Frag16 qf[3];
  {
    const unsigned short* qb = qp + (size_t)(qrow0 + cn) * (NH * HDQ) + h * HDQ + kb8;
    #pragma unroll
    for (int c = 0; c < 3; c++) {
      qf[c].u[0] = *(const uint4*)(qb + c * 32);
      qf[c].u[1] = *(const uint4*)(qb + c * 32 + 16);
    }
  }
  const unsigned short* vph = vp + (size_t)h * (HDV * L_SEQ);

  v8f O[5]; float m8[8], l8[8];
  for (int t = 0; t < 5; t++) for (int r = 0; r < 8; r++) O[t][r] = 0.f;
  for (int r = 0; r < 8; r++) { m8[r] = -1e30f; l8[r] = 0.f; }
  const float scl = 0.11785113019775793f;   // HD^-0.5, HD=72

  for (int k0 = kbeg; k0 < kend; k0 += 32) {
    __syncthreads();
    // K tile: async DMA, 32 x 96 bf16
    for (int i = tid; i < 32 * 12; i += 256) {
      int r = i / 12, c = (i % 12) << 3;
      async_copy16(&kt[r * HDQ + c],
                   kp + (size_t)(k0 + r) * (NH * HDQ) + h * HDQ + c);
    }
    // V tile: async DMA, 80 rows (hd) x 32 keys bf16 (already transposed)
    for (int i = tid; i < HDV * 4; i += 256) {
      int r = i >> 2, c = (i & 3) << 3;
      async_copy16(&vt[r * 32 + c], vph + (size_t)r * L_SEQ + k0 + c);
    }
    wait_async0();
    __syncthreads();

    v8f s0, s1;
    for (int r = 0; r < 8; r++) { s0[r] = 0.f; s1[r] = 0.f; }
    #pragma unroll
    for (int c = 0; c < 3; c++) {
      Frag16 b0, b1;
      int off0 = cn * HDQ + c * 32 + kb8;
      b0.u[0] = *(const uint4*)&kt[off0];
      b0.u[1] = *(const uint4*)&kt[off0 + 16];
      int off1 = (16 + cn) * HDQ + c * 32 + kb8;
      b1.u[0] = *(const uint4*)&kt[off1];
      b1.u[1] = *(const uint4*)&kt[off1 + 16];
      s0 = __builtin_amdgcn_wmma_f32_16x16x32_bf16(false, qf[c].v, false, b0.v,
                                                   (short)0, s0, false, false);
      s1 = __builtin_amdgcn_wmma_f32_16x16x32_bf16(false, qf[c].v, false, b1.v,
                                                   (short)0, s1, false, false);
    }

    float alpha[8];
    #pragma unroll
    for (int r = 0; r < 8; r++) {
      float a0 = s0[r] * scl, a1 = s1[r] * scl;
      float mx = fmaxf(a0, a1);
      mx = fmaxf(mx, __shfl_xor(mx, 1, 32));
      mx = fmaxf(mx, __shfl_xor(mx, 2, 32));
      mx = fmaxf(mx, __shfl_xor(mx, 4, 32));
      mx = fmaxf(mx, __shfl_xor(mx, 8, 32));
      float mn = fmaxf(m8[r], mx);
      float al = __expf(m8[r] - mn);
      float p0 = __expf(a0 - mn), p1 = __expf(a1 - mn);
      float rs = p0 + p1;
      rs += __shfl_xor(rs, 1, 32);
      rs += __shfl_xor(rs, 2, 32);
      rs += __shfl_xor(rs, 4, 32);
      rs += __shfl_xor(rs, 8, 32);
      m8[r] = mn; l8[r] = l8[r] * al + rs; alpha[r] = al;
      s0[r] = p0; s1[r] = p1;
    }
    #pragma unroll
    for (int t = 0; t < 5; t++)
      #pragma unroll
      for (int r = 0; r < 8; r++) O[t][r] *= alpha[r];

    {
      unsigned short* pw = &pb[w * 512];
      #pragma unroll
      for (int r = 0; r < 8; r++) {
        int row = r + 8 * hi;
        pw[row * 32 + cn]      = (unsigned short)f2bf(s0[r]);
        pw[row * 32 + 16 + cn] = (unsigned short)f2bf(s1[r]);
      }
    }
    __syncthreads();

    Frag16 pa;
    int poff = w * 512 + cn * 32 + kb8;
    pa.u[0] = *(const uint4*)&pb[poff];
    pa.u[1] = *(const uint4*)&pb[poff + 16];
    #pragma unroll
    for (int t = 0; t < 5; t++) {
      Frag16 bv;
      int voff = (t * 16 + cn) * 32 + kb8;
      bv.u[0] = *(const uint4*)&vt[voff];
      bv.u[1] = *(const uint4*)&vt[voff + 16];
      O[t] = __builtin_amdgcn_wmma_f32_16x16x32_bf16(false, pa.v, false, bv.v,
                                                     (short)0, O[t], false, false);
    }
  }

  float il[8];
  for (int r = 0; r < 8; r++) il[r] = 1.f / l8[r];
  #pragma unroll
  for (int t = 0; t < 5; t++) {
    int d = t * 16 + cn;
    if (d < HD) {
      #pragma unroll
      for (int r = 0; r < 8; r++) {
        int row = qrow0 + r + 8 * hi;
        out[(size_t)row * HID + h * HD + d] = O[t][r] * il[r];
      }
    }
  }
}

// ---------------------------------------------------------------- launcher
extern "C" void kernel_launch(void* const* d_in, const int* in_sizes, int n_in,
                              void* d_out, int out_size, void* d_ws, size_t ws_size,
                              hipStream_t stream) {
  const float* hidden = (const float*)d_in[0];
  const float* rot    = (const float*)d_in[1];
  const float* qkv_w  = (const float*)d_in[2];
  const float* qkv_b  = (const float*)d_in[3];
  const float* proj_w = (const float*)d_in[4];
  const float* proj_b = (const float*)d_in[5];
  const float* fc1_w  = (const float*)d_in[6];
  const float* fc1_b  = (const float*)d_in[7];
  const float* fc2_w  = (const float*)d_in[8];
  const float* fc2_b  = (const float*)d_in[9];
  const float* ln1s   = (const float*)d_in[10];
  const float* ln1b   = (const float*)d_in[11];
  const float* ln2s   = (const float*)d_in[12];
  const float* ln2b   = (const float*)d_in[13];
  const int*   cu     = (const int*)d_in[14];
  const int nseg = in_sizes[14] - 1;
  float* outp = (float*)d_out;

  char* ws = (char*)d_ws;
  size_t off = 0;
  auto alloc = [&](size_t b) { size_t o = off; off = (off + b + 255) & ~(size_t)255; return o; };
  float*          lnbuf  = (float*)(ws + alloc((size_t)L_SEQ * HID * 4));  // LN1 out, reused for LN2 out
  unsigned short* qkvWt  = (unsigned short*)(ws + alloc((size_t)QKV_N * HID * 2));
  unsigned short* projWt = (unsigned short*)(ws + alloc((size_t)HID * HID * 2));
  unsigned short* fc1Wt  = (unsigned short*)(ws + alloc((size_t)FFP * HID * 2));
  unsigned short* fc2Wt  = (unsigned short*)(ws + alloc((size_t)HID * FFP * 2));
  float*          big    = (float*)(ws + alloc((size_t)L_SEQ * FFP * 4));  // qkv out, reused for fc1g
  unsigned short* qpb    = (unsigned short*)(ws + alloc((size_t)L_SEQ * NH * HDQ * 2));
  unsigned short* kpb    = (unsigned short*)(ws + alloc((size_t)L_SEQ * NH * HDQ * 2));
  unsigned short* vpb    = (unsigned short*)(ws + alloc((size_t)NH * HDV * L_SEQ * 2));
  float*          attn_o = (float*)(ws + alloc((size_t)L_SEQ * HID * 4));
  float*          h1     = (float*)(ws + alloc((size_t)L_SEQ * HID * 4));

  // LN1
  ln_kernel<<<L_SEQ, 256, 0, stream>>>(hidden, ln1s, ln1b, lnbuf);

  // weight transpose + bf16 convert (zero-padded)
  auto cgrid = [](long long n) { return (unsigned)((n + 255) / 256); };
  convT_kernel<<<cgrid((long long)QKV_N * HID), 256, 0, stream>>>(qkv_w, qkvWt, HID, QKV_N, HID, QKV_N);
  convT_kernel<<<cgrid((long long)HID * HID),   256, 0, stream>>>(proj_w, projWt, HID, HID, HID, HID);
  convT_kernel<<<cgrid((long long)FFP * HID),   256, 0, stream>>>(fc1_w, fc1Wt, HID, FF, HID, FFP);
  convT_kernel<<<cgrid((long long)HID * FFP),   256, 0, stream>>>(fc2_w, fc2Wt, FF, HID, FFP, HID);

  // QKV = LN1 @ qkv_w + b
  gemm_bf16<0, 0><<<dim3(QKV_N / 128, L_SEQ / 128), 256, 0, stream>>>(
      lnbuf, HID, qkvWt, qkv_b, nullptr, 0, big, QKV_N, QKV_N, HID);

  // RoPE + pack q/k (padded) and v (transposed) to bf16
  rope_kernel<<<(L_SEQ * NH * HDQ) / 256, 256, 0, stream>>>(big, rot, qpb, kpb, vpb);

  // block-diagonal flash attention
  attn_kernel<<<dim3(L_SEQ / 128, NH), 256, 0, stream>>>(qpb, kpb, vpb, cu, nseg, attn_o);

  // h1 = hidden + attn_o @ proj_w + b
  gemm_bf16<0, 1><<<dim3(HID / 128, L_SEQ / 128), 256, 0, stream>>>(
      attn_o, HID, projWt, proj_b, hidden, HID, h1, HID, HID, HID);

  // LN2
  ln_kernel<<<L_SEQ, 256, 0, stream>>>(h1, ln2s, ln2b, lnbuf);

  // fc1g = gelu(LN2 @ fc1_w + b)  (stored padded to FFP with zeros)
  gemm_bf16<1, 0><<<dim3(FFP / 128, L_SEQ / 128), 256, 0, stream>>>(
      lnbuf, HID, fc1Wt, fc1_b, nullptr, 0, big, FFP, FF, HID);

  // out = h1 + fc1g @ fc2_w + b
  gemm_bf16<0, 1><<<dim3(HID / 128, L_SEQ / 128), 256, 0, stream>>>(
      big, FFP, fc2Wt, fc2_b, h1, HID, outp, HID, HID, FFP);
}
